// SparseMoELayer_62062277427624
// MI455X (gfx1250) — matrix operand order repaired and checked by reading, not compiled, not run on hardware
//
#include <hip/hip_runtime.h>
#include <hip/hip_bf16.h>

#define H_DIM 1024
#define E_NUM 8
#define I_DIM 4096
#define T_NUM 4096   // B*S = 2*2048

typedef __attribute__((ext_vector_type(16))) __bf16 v16bf;
typedef __attribute__((ext_vector_type(8)))  float  v8f;

struct __align__(16) U4 { unsigned int a, b, c, d; };

union Frag16 {
    v16bf  v;
    U4     q[2];
    __bf16 h[16];
};

// ---------------------------------------------------------------------------
// Zero output + expert counters
// ---------------------------------------------------------------------------
struct __align__(16) F4 { float a, b, c, d; };

__global__ __launch_bounds__(256) void moe_zero(float* __restrict__ out,
                                                int* __restrict__ counts,
                                                int n4) {
    int i = blockIdx.x * blockDim.x + threadIdx.x;
    if (i < n4) {
        F4 z; z.a = 0.f; z.b = 0.f; z.c = 0.f; z.d = 0.f;
        reinterpret_cast<F4*>(out)[i] = z;
    }
    if (i < E_NUM) counts[i] = 0;
}

// ---------------------------------------------------------------------------
// fp32 -> bf16 (straight copy, for activations x)
// ---------------------------------------------------------------------------
__global__ __launch_bounds__(256) void moe_cvt(const float* __restrict__ s,
                                               __bf16* __restrict__ d, int n) {
    int i = (blockIdx.x * blockDim.x + threadIdx.x) * 4;
    if (i < n) {
        d[i + 0] = (__bf16)s[i + 0];
        d[i + 1] = (__bf16)s[i + 1];
        d[i + 2] = (__bf16)s[i + 2];
        d[i + 3] = (__bf16)s[i + 3];
    }
}

// ---------------------------------------------------------------------------
// fp32 [E][K][N] -> bf16 [E][N][K] (transpose at convert time so the GEMM
// B-tile staging is straight b128 copies; 64x64 LDS-tiled transpose)
// ---------------------------------------------------------------------------
__global__ __launch_bounds__(256) void moe_cvt_t(const float* __restrict__ s,
                                                 __bf16* __restrict__ d,
                                                 int K, int N) {
    __shared__ __bf16 tile[64][72];
    const int e  = blockIdx.z;
    const int k0 = blockIdx.y * 64;
    const int n0 = blockIdx.x * 64;
    const float* se = s + (size_t)e * K * N;
    __bf16*      de = d + (size_t)e * N * K;

    const int tid = threadIdx.x;
    const int tr  = tid >> 2;        // 0..63
    const int tc  = (tid & 3) * 16;  // 0,16,32,48

    const float* srow = se + (size_t)(k0 + tr) * N + n0 + tc;
#pragma unroll
    for (int j = 0; j < 16; ++j) tile[tr][tc + j] = (__bf16)srow[j];
    __syncthreads();

    __bf16* drow = de + (size_t)(n0 + tr) * K + k0 + tc;
#pragma unroll
    for (int j = 0; j < 16; ++j) drow[j] = tile[tc + j][tr];
}

// ---------------------------------------------------------------------------
// Router: one wave (32 lanes) per token. logits -> softmax -> top2 -> renorm,
// then atomic append into per-expert buckets.
// ---------------------------------------------------------------------------
__global__ __launch_bounds__(256) void moe_router(const float* __restrict__ x,
                                                  const float* __restrict__ gw,
                                                  const float* __restrict__ gb,
                                                  int*   __restrict__ counts,
                                                  int*   __restrict__ btok,
                                                  float* __restrict__ bwgt) {
    const int tid  = threadIdx.x;
    const int lane = tid & 31;
    const int wid  = tid >> 5;
    const int t    = blockIdx.x * 8 + wid;

    const float* xrow = x + (size_t)t * H_DIM;
    float acc[E_NUM];
#pragma unroll
    for (int e = 0; e < E_NUM; ++e) acc[e] = 0.f;

    for (int it = 0; it < H_DIM / 32; ++it) {
        const int h = it * 32 + lane;
        const float xv = xrow[h];
        const float* g = gw + (size_t)h * E_NUM;
#pragma unroll
        for (int e = 0; e < E_NUM; ++e) acc[e] += xv * g[e];
    }
#pragma unroll
    for (int e = 0; e < E_NUM; ++e) {
#pragma unroll
        for (int off = 16; off; off >>= 1)
            acc[e] += __shfl_xor(acc[e], off, 32);
    }
    if (lane == 0) {
        float l[E_NUM], mx = -1e30f;
#pragma unroll
        for (int e = 0; e < E_NUM; ++e) { l[e] = acc[e] + gb[e]; mx = fmaxf(mx, l[e]); }
        float p[E_NUM], s = 0.f;
#pragma unroll
        for (int e = 0; e < E_NUM; ++e) { p[e] = __expf(l[e] - mx); s += p[e]; }
        const float inv = 1.0f / s;
#pragma unroll
        for (int e = 0; e < E_NUM; ++e) p[e] *= inv;

        int i0 = 0;
#pragma unroll
        for (int e = 1; e < E_NUM; ++e) if (p[e] > p[i0]) i0 = e;
        int i1 = (i0 == 0) ? 1 : 0;
#pragma unroll
        for (int e = 0; e < E_NUM; ++e) if (e != i0 && p[e] > p[i1]) i1 = e;

        float w0 = p[i0], w1 = p[i1];
        const float dn = 1.0f / (w0 + w1 + 1e-6f);
        w0 *= dn; w1 *= dn;

        int s0 = atomicAdd(&counts[i0], 1);
        btok[i0 * T_NUM + s0] = t; bwgt[i0 * T_NUM + s0] = w0;
        int s1 = atomicAdd(&counts[i1], 1);
        btok[i1 * T_NUM + s1] = t; bwgt[i1 * T_NUM + s1] = w1;
    }
}

// ---------------------------------------------------------------------------
// 8-entry exclusive prefix sum (hmid row offsets per expert)
// ---------------------------------------------------------------------------
__global__ void moe_offsets(const int* __restrict__ counts,
                            int* __restrict__ offsets) {
    if (threadIdx.x == 0) {
        int a = 0;
        for (int e = 0; e < E_NUM; ++e) { offsets[e] = a; a += counts[e]; }
    }
}

// ---------------------------------------------------------------------------
// Grouped GEMM, 128x128 macro tile, BK=32, 8 waves of 32x64 WMMA tiles.
// Weights pre-transposed: WT is [E][N][K]. Tiles staged with CDNA5 async
// global->LDS DMA (GLOBAL_LOAD_ASYNC_TO_LDS_B128, ASYNCcnt), double-buffered,
// one s_wait_asynccnt + barrier per K-step.
// MODE 0: hmid = gelu(gather(xb) @ W1[e] + b1[e]) , K=H, N=I
// MODE 1: out  += gate_w * (hmid @ W2[e] + b2[e]) , K=I, N=H (atomic scatter)
// ---------------------------------------------------------------------------
template <int MODE>
__global__ __launch_bounds__(256) void moe_gemm(
    const __bf16* __restrict__ A,      // MODE0: xb [T,H]    MODE1: hmid [2T,I]
    const __bf16* __restrict__ WT,     // MODE0: W1T [E,I,H] MODE1: W2T [E,H,I]
    const float*  __restrict__ bias,   // [E,N]
    const int*    __restrict__ counts,
    const int*    __restrict__ offsets,
    const int*    __restrict__ btok,
    const float*  __restrict__ bwgt,
    __bf16*       __restrict__ hmid,
    float*        __restrict__ outp,
    int K, int N) {
    constexpr int BM = 128, BN = 128, BK = 32, LDW = BK + 8;
    __shared__ __align__(16) __bf16 As[2][BM][LDW];
    __shared__ __align__(16) __bf16 Bt[2][BN][LDW];

    const int e    = blockIdx.z;
    const int cnt  = counts[e];
    const int row0 = blockIdx.y * BM;
    if (row0 >= cnt) return;
    const int off = offsets[e];
    const int n0  = blockIdx.x * BN;

    const int tid  = threadIdx.x;
    const int lane = tid & 31;
    const int wid  = tid >> 5;
    const int wm   = (wid & 3) * 32;   // wave M offset in tile
    const int wn   = (wid >> 2) * 64;  // wave N offset in tile

    // cooperative staging coordinates (same pattern for A and Bt)
    const int lrow  = tid >> 1;         // 0..127
    const int lhalf = (tid & 1) * 16;   // element offset 0 / 16

    // A source row (gather for MODE 0, packed for MODE 1)
    const __bf16* arow_ptr;
    {
        const int r = row0 + lrow;
        if (MODE == 0) {
            const int tok = (r < cnt) ? btok[e * T_NUM + r] : btok[e * T_NUM];
            arow_ptr = A + (size_t)tok * K;
        } else {
            const int ar = off + ((r < cnt) ? r : 0);
            arow_ptr = A + (size_t)ar * K;
        }
    }
    // B source row: WT row (n0 + lrow), contiguous K
    const __bf16* brow_ptr = WT + ((size_t)e * N + n0 + lrow) * K;

    // CDNA5 async global->LDS staging: per-lane 2x16B for A tile, 2x16B for
    // B tile. INST_OFFSET applies to both LDS and global sides, so base+16
    // covers the second 16B chunk of each 32B row segment.
    auto async_stage = [&](int buf, int k0) {
        const unsigned la = (unsigned)(unsigned long long)&As[buf][lrow][lhalf];
        const unsigned lb = (unsigned)(unsigned long long)&Bt[buf][lrow][lhalf];
        const __bf16* ga  = arow_ptr + k0 + lhalf;
        const __bf16* gbp = brow_ptr + k0 + lhalf;
        asm volatile(
            "global_load_async_to_lds_b128 %0, %2, off\n\t"
            "global_load_async_to_lds_b128 %0, %2, off offset:16\n\t"
            "global_load_async_to_lds_b128 %1, %3, off\n\t"
            "global_load_async_to_lds_b128 %1, %3, off offset:16"
            :
            : "v"(la), "v"(lb), "v"(ga), "v"(gbp)
            : "memory");
        if (k0 + 2 * BK <= K - BK)
            __builtin_prefetch(brow_ptr + k0 + 2 * BK, 0, 1);
    };

    v8f acc[2][4] = {};
    async_stage(0, 0);

    const int KT = K / BK;
    for (int kt = 0; kt < KT; ++kt) {
        const int buf = kt & 1;
        asm volatile("s_wait_asynccnt 0x0" ::: "memory");  // my DMA into buf done
        __syncthreads();                                   // everyone's DMA done
        if (kt + 1 < KT) async_stage(buf ^ 1, (kt + 1) * BK);

        // fragments per documented 16-bit WMMA VGPR layouts
        const int m  = lane & 15;
        const int kA = (lane >> 4) * 8;    // A: lanes<16 K{0..7,16..23}, else {8..15,24..31}
        const int kB = (lane >> 4) * 8;    // B: lanes<16 K{0..7,16..23}, else {8..15,24..31}

        Frag16 a0, a1, bf[4];
        a0.q[0] = *reinterpret_cast<const U4*>(&As[buf][wm + m][kA]);
        a0.q[1] = *reinterpret_cast<const U4*>(&As[buf][wm + m][kA + 16]);
        a1.q[0] = *reinterpret_cast<const U4*>(&As[buf][wm + 16 + m][kA]);
        a1.q[1] = *reinterpret_cast<const U4*>(&As[buf][wm + 16 + m][kA + 16]);
#pragma unroll
        for (int ni = 0; ni < 4; ++ni) {
            bf[ni].q[0] = *reinterpret_cast<const U4*>(&Bt[buf][wn + ni * 16 + m][kB]);
            bf[ni].q[1] = *reinterpret_cast<const U4*>(&Bt[buf][wn + ni * 16 + m][kB + 16]);
        }
#pragma unroll
        for (int ni = 0; ni < 4; ++ni) {
            acc[0][ni] = __builtin_amdgcn_wmma_f32_16x16x32_bf16(
                false, a0.v, false, bf[ni].v, (short)0, acc[0][ni], false, false);
            acc[1][ni] = __builtin_amdgcn_wmma_f32_16x16x32_bf16(
                false, a1.v, false, bf[ni].v, (short)0, acc[1][ni], false, false);
        }
    }

    // epilogue (C layout: lanes<16 -> M=r, lanes>=16 -> M=8+r; N=lane&15)
    const int mhalf = (lane >> 4) * 8;
    const int ncol  = lane & 15;
#pragma unroll
    for (int mi = 0; mi < 2; ++mi) {
#pragma unroll
        for (int ni = 0; ni < 4; ++ni) {
            const int colg = n0 + wn + ni * 16 + ncol;
            const float bv = bias[(size_t)e * N + colg];
#pragma unroll
            for (int r = 0; r < 8; ++r) {
                const int mrow = row0 + wm + mi * 16 + mhalf + r;
                if (mrow < cnt) {
                    float v = acc[mi][ni][r] + bv;
                    if (MODE == 0) {
                        const float u = 0.7978845608028654f * (v + 0.044715f * v * v * v);
                        v = 0.5f * v * (1.0f + tanhf(u));
                        hmid[(size_t)(off + mrow) * N + colg] = (__bf16)v;
                    } else {
                        const int   tok = btok[e * T_NUM + mrow];
                        const float w   = bwgt[e * T_NUM + mrow];
                        atomicAdd(&outp[(size_t)tok * H_DIM + colg], w * v);
                    }
                }
            }
        }
    }
}

// ---------------------------------------------------------------------------
extern "C" void kernel_launch(void* const* d_in, const int* in_sizes, int n_in,
                              void* d_out, int out_size, void* d_ws, size_t ws_size,
                              hipStream_t stream) {
    (void)in_sizes; (void)n_in; (void)out_size; (void)ws_size;
    const float* x  = (const float*)d_in[0];
    const float* gw = (const float*)d_in[1];
    const float* gb = (const float*)d_in[2];
    const float* W1 = (const float*)d_in[3];
    const float* b1 = (const float*)d_in[4];
    const float* W2 = (const float*)d_in[5];
    const float* b2 = (const float*)d_in[6];
    float* out = (float*)d_out;

    char* ws = (char*)d_ws;
    size_t o = 0;
    auto alloc = [&](size_t bytes) -> void* {
        void* p = ws + o;
        o += (bytes + 255) & ~(size_t)255;
        return p;
    };
    __bf16* xb     = (__bf16*)alloc((size_t)T_NUM * H_DIM * 2);
    __bf16* W1T    = (__bf16*)alloc((size_t)E_NUM * H_DIM * I_DIM * 2); // [E][I][H]
    __bf16* W2T    = (__bf16*)alloc((size_t)E_NUM * I_DIM * H_DIM * 2); // [E][H][I]
    int*    counts = (int*)  alloc(64);
    int*    offs   = (int*)  alloc(64);
    int*    btok   = (int*)  alloc((size_t)E_NUM * T_NUM * 4);
    float*  bwgt   = (float*)alloc((size_t)E_NUM * T_NUM * 4);
    __bf16* hmid   = (__bf16*)alloc((size_t)2 * T_NUM * I_DIM * 2);

    const int nOut4 = (T_NUM * H_DIM) / 4;
    moe_zero<<<(nOut4 + 255) / 256, 256, 0, stream>>>(out, counts, nOut4);

    const int nX = T_NUM * H_DIM;
    moe_cvt<<<(nX / 4 + 255) / 256, 256, 0, stream>>>(x, xb, nX);
    // W1 [E][H][I] -> W1T [E][I][H] : K=H, N=I
    moe_cvt_t<<<dim3(I_DIM / 64, H_DIM / 64, E_NUM), 256, 0, stream>>>(W1, W1T, H_DIM, I_DIM);
    // W2 [E][I][H] -> W2T [E][H][I] : K=I, N=H
    moe_cvt_t<<<dim3(H_DIM / 64, I_DIM / 64, E_NUM), 256, 0, stream>>>(W2, W2T, I_DIM, H_DIM);

    moe_router<<<T_NUM / 8, 256, 0, stream>>>(x, gw, gb, counts, btok, bwgt);
    moe_offsets<<<1, 32, 0, stream>>>(counts, offs);

    // GEMM1: [cnt_e, H] x [H, I] -> hmid ; worst case 32 row-tiles per expert
    moe_gemm<0><<<dim3(I_DIM / 128, T_NUM / 128, E_NUM), 256, 0, stream>>>(
        xb, W1T, b1, counts, offs, btok, bwgt, hmid, out, H_DIM, I_DIM);
    // GEMM2: [cnt_e, I] x [I, H] -> gated atomic scatter into out
    moe_gemm<1><<<dim3(H_DIM / 128, T_NUM / 128, E_NUM), 256, 0, stream>>>(
        hmid, W2T, b2, counts, offs, btok, bwgt, hmid, out, I_DIM, H_DIM);
}
